// CausalSelfAttention_19061064860284
// MI455X (gfx1250) — matrix operand order repaired and checked by reading, not compiled, not run on hardware
//
#include <hip/hip_runtime.h>

typedef __bf16 bf16_t;
typedef __attribute__((ext_vector_type(16))) __bf16 v16bf;
typedef __attribute__((ext_vector_type(8)))  __bf16 v8bf;
typedef __attribute__((ext_vector_type(8)))  float   v8f;

#define BB 4
#define LL 2048
#define DD 768
#define HH 12
#define HD 64
#define TD 2304           // 3*D
#define MM (BB*LL)        // 8192

__device__ __forceinline__ v8f wmma_bf16(v16bf a, v16bf b, v8f c) {
  return __builtin_amdgcn_wmma_f32_16x16x32_bf16(false, a, false, b, (short)0, c, false, false);
}

// ---------------- conversion kernels ----------------

__global__ void k_f32_to_bf16(const float* __restrict__ in, bf16_t* __restrict__ out, int n) {
  int i = blockIdx.x * blockDim.x + threadIdx.x;
  if (i < n) out[i] = (bf16_t)in[i];
}

// in[rows][cols] (f32) -> out[cols][rows] (bf16)
__global__ void k_transpose_f32_to_bf16(const float* __restrict__ in, bf16_t* __restrict__ out,
                                        int rows, int cols) {
  int c = blockIdx.x * blockDim.x + threadIdx.x;
  int r = blockIdx.y * blockDim.y + threadIdx.y;
  if (c < cols && r < rows)
    out[(size_t)c * rows + r] = (bf16_t)in[(size_t)r * cols + c];
}

// vt[b][h][d][l] = qkv[(b*L+l)][2*D + h*HD + d]
__global__ void k_build_vt(const bf16_t* __restrict__ qkv, bf16_t* __restrict__ vt) {
  int i = blockIdx.x * blockDim.x + threadIdx.x;
  if (i >= BB * HH * HD * LL) return;
  int l = i % LL;
  int d = (i / LL) % HD;
  int h = (i / (LL * HD)) % HH;
  int b = i / (LL * HD * HH);
  vt[i] = qkv[(size_t)(b * LL + l) * TD + 2 * DD + h * HD + d];
}

// ------- WMMA GEMM, 16(M)x64(N) per wave: C = A[M,K] * Bt[N,K]^T -------
// A fragment loaded once per K-step, reused across 4 B fragments (4x intensity).

__device__ __forceinline__ v16bf load_afrag(const bf16_t* arow, int kb, int half) {
  v8bf lo = *(const v8bf*)(arow + kb + 16 * half);
  v8bf hi = *(const v8bf*)(arow + kb + 16 * half + 8);
  v16bf a;
#pragma unroll
  for (int e = 0; e < 8; ++e) { a[e] = lo[e]; a[e + 8] = hi[e]; }
  return a;
}

__global__ void k_gemm_bf16(const bf16_t* __restrict__ A, const bf16_t* __restrict__ Bt,
                            bf16_t* __restrict__ C, int M, int N, int K) {
  const int lane = threadIdx.x & 31;
  const int wave = threadIdx.x >> 5;
  const int half = lane >> 4;
  const int lr   = lane & 15;
  const int n0 = blockIdx.x * 64;
  const int m0 = blockIdx.y * 64 + wave * 16;

  const bf16_t* arow = A + (size_t)(m0 + lr) * K;
  v8f acc[4];
#pragma unroll
  for (int t = 0; t < 4; ++t) acc[t] = (v8f){};

  for (int kb = 0; kb < K; kb += 32) {
    v16bf a = load_afrag(arow, kb, half);
#pragma unroll
    for (int t = 0; t < 4; ++t) {
      v16bf b = *(const v16bf*)(Bt + (size_t)(n0 + 16 * t + lr) * K + kb + 16 * half);
      acc[t] = wmma_bf16(a, b, acc[t]);
    }
  }
  bf16_t* crow = C + (size_t)(m0 + 8 * half) * N + n0 + lr;
#pragma unroll
  for (int t = 0; t < 4; ++t)
#pragma unroll
    for (int r = 0; r < 8; ++r)
      crow[(size_t)r * N + 16 * t] = (bf16_t)acc[t][r];
}

__global__ void k_gemm_f32_bias(const bf16_t* __restrict__ A, const bf16_t* __restrict__ Bt,
                                const float* __restrict__ bias, float* __restrict__ C,
                                int M, int N, int K) {
  const int lane = threadIdx.x & 31;
  const int wave = threadIdx.x >> 5;
  const int half = lane >> 4;
  const int lr   = lane & 15;
  const int n0 = blockIdx.x * 64;
  const int m0 = blockIdx.y * 64 + wave * 16;

  const bf16_t* arow = A + (size_t)(m0 + lr) * K;
  v8f acc[4];
#pragma unroll
  for (int t = 0; t < 4; ++t) acc[t] = (v8f){};

  for (int kb = 0; kb < K; kb += 32) {
    v16bf a = load_afrag(arow, kb, half);
#pragma unroll
    for (int t = 0; t < 4; ++t) {
      v16bf b = *(const v16bf*)(Bt + (size_t)(n0 + 16 * t + lr) * K + kb + 16 * half);
      acc[t] = wmma_bf16(a, b, acc[t]);
    }
  }
  float* crow = C + (size_t)(m0 + 8 * half) * N + n0 + lr;
#pragma unroll
  for (int t = 0; t < 4; ++t) {
    float bv = bias[n0 + 16 * t + lr];
#pragma unroll
    for (int r = 0; r < 8; ++r)
      crow[(size_t)r * N + 16 * t] = acc[t][r] + bv;
  }
}

// ---------------- flash attention (causal, online softmax) ----------------
// qkv: [B*L, 3*D] bf16 (q | k | v blocks of D, head-major within block)
// vt : [B, H, HD, L] bf16
// out: [B*L, D] bf16
__global__ void k_flash_attn(const bf16_t* __restrict__ qkv,
                             const bf16_t* __restrict__ vt,
                             bf16_t* __restrict__ out) {
  __shared__ alignas(32) bf16_t pbuf[4][16][40];  // 80B row stride (16B-aligned runs)

  const int lane = threadIdx.x & 31;
  const int wave = threadIdx.x >> 5;
  const int half = lane >> 4;
  const int lr   = lane & 15;
  const int h = blockIdx.y;
  const int b = blockIdx.z;
  const int q0 = blockIdx.x * 64 + wave * 16;

  const bf16_t* qbase = qkv + (size_t)(b * LL + q0 + lr) * TD + h * HD;
  const bf16_t* kbase = qkv + (size_t)(b * LL) * TD + DD + h * HD;
  const bf16_t* vbase = vt + (size_t)((b * HH + h) * HD) * LL;

  // Q fragments, hd chunks [0,32) and [32,64); loop-invariant
  v16bf aq[2];
#pragma unroll
  for (int c = 0; c < 2; ++c) {
    v8bf lo = *(const v8bf*)(qbase + 32 * c + 16 * half);
    v8bf hi = *(const v8bf*)(qbase + 32 * c + 16 * half + 8);
#pragma unroll
    for (int e = 0; e < 8; ++e) { aq[c][e] = lo[e]; aq[c][e + 8] = hi[e]; }
  }

  float mrow[8], lrow[8];
  v8f o[4];
#pragma unroll
  for (int r = 0; r < 8; ++r) { mrow[r] = -3.0e38f; lrow[r] = 0.0f; }
#pragma unroll
  for (int t = 0; t < 4; ++t) o[t] = (v8f){};

  const int kend = q0 + 15;              // last valid key for this wave's rows
  for (int kb = 0; kb <= kend; kb += 32) {
    // S = Q K^T for two 16-key subtiles
    v8f s[2];
#pragma unroll
    for (int j = 0; j < 2; ++j) {
      const int kb2 = kb + 16 * j;
      v8f acc = {};
#pragma unroll
      for (int c = 0; c < 2; ++c) {
        v16bf bk = *(const v16bf*)(kbase + (size_t)(kb2 + lr) * TD + 32 * c + 16 * half);
        acc = wmma_bf16(aq[c], bk, acc);
      }
      s[j] = acc;
    }
    // scale + causal mask
#pragma unroll
    for (int j = 0; j < 2; ++j) {
#pragma unroll
      for (int r = 0; r < 8; ++r) {
        const int mg = q0 + r + 8 * half;
        const int kg = kb + 16 * j + lr;
        const float v = s[j][r] * 0.125f;           // 1/sqrt(64)
        s[j][r] = (kg <= mg) ? v : -3.0e38f;
      }
    }
    // online softmax row statistics (rows live across 16 lanes of each half)
    float alpha[8];
#pragma unroll
    for (int r = 0; r < 8; ++r) {
      float mx = fmaxf(s[0][r], s[1][r]);
      mx = fmaxf(mx, __shfl_xor(mx, 1, 32));
      mx = fmaxf(mx, __shfl_xor(mx, 2, 32));
      mx = fmaxf(mx, __shfl_xor(mx, 4, 32));
      mx = fmaxf(mx, __shfl_xor(mx, 8, 32));
      const float mnew = fmaxf(mrow[r], mx);
      alpha[r] = __expf(mrow[r] - mnew);
      mrow[r] = mnew;
      const float p0 = __expf(s[0][r] - mnew);
      const float p1 = __expf(s[1][r] - mnew);
      s[0][r] = p0; s[1][r] = p1;
      float ps = p0 + p1;
      ps += __shfl_xor(ps, 1, 32);
      ps += __shfl_xor(ps, 2, 32);
      ps += __shfl_xor(ps, 4, 32);
      ps += __shfl_xor(ps, 8, 32);
      lrow[r] = lrow[r] * alpha[r] + ps;
    }
    // rescale running output
#pragma unroll
    for (int t = 0; t < 4; ++t)
#pragma unroll
      for (int r = 0; r < 8; ++r) o[t][r] *= alpha[r];

    // stage P (16x32) to LDS in row-major bf16, re-read as A fragment
#pragma unroll
    for (int j = 0; j < 2; ++j)
#pragma unroll
      for (int r = 0; r < 8; ++r)
        pbuf[wave][r + 8 * half][16 * j + lr] = (bf16_t)s[j][r];

    v16bf ap;
    {
      v8bf lo = *(const v8bf*)(&pbuf[wave][lr][16 * half]);
      v8bf hi = *(const v8bf*)(&pbuf[wave][lr][16 * half + 8]);
#pragma unroll
      for (int e = 0; e < 8; ++e) { ap[e] = lo[e]; ap[e + 8] = hi[e]; }
    }
    // O += P @ V : 4 hd tiles of 16
#pragma unroll
    for (int t = 0; t < 4; ++t) {
      v16bf bv = *(const v16bf*)(vbase + (size_t)(16 * t + lr) * LL + kb + 16 * half);
      o[t] = wmma_bf16(ap, bv, o[t]);
    }
  }

  // normalize + store
  bf16_t* obase = out + (size_t)(b * LL + q0 + 8 * half) * DD + h * HD + lr;
#pragma unroll
  for (int r = 0; r < 8; ++r) {
    const float inv = 1.0f / lrow[r];
#pragma unroll
    for (int t = 0; t < 4; ++t)
      obase[(size_t)r * DD + 16 * t] = (bf16_t)(o[t][r] * inv);
  }
}

// ---------------- launcher ----------------

extern "C" void kernel_launch(void* const* d_in, const int* in_sizes, int n_in,
                              void* d_out, int out_size, void* d_ws, size_t ws_size,
                              hipStream_t stream) {
  const float* x      = (const float*)d_in[0];   // [B*L, D]
  const float* w_attn = (const float*)d_in[1];   // [D, 3D]
  const float* w_proj = (const float*)d_in[2];   // [D, D]
  const float* b_proj = (const float*)d_in[3];   // [D]
  float* outp = (float*)d_out;

  char* ws = (char*)d_ws;
  bf16_t* xb   = (bf16_t*)(ws);                         // 8192*768*2   = 12,582,912
  bf16_t* wat  = (bf16_t*)(ws + 12582912);              // 2304*768*2   =  3,538,944
  bf16_t* wpt  = (bf16_t*)(ws + 16121856);              //  768*768*2   =  1,179,648
  bf16_t* qkvb = (bf16_t*)(ws + 17301504);              // 8192*2304*2  = 37,748,736
  bf16_t* vtb  = (bf16_t*)(ws + 55050240);              // 4*12*64*2048*2 = 12,582,912
  bf16_t* ao   = (bf16_t*)(ws + 67633152);              // 8192*768*2   = 12,582,912
  // total: 80,216,064 bytes

  // 1) convert inputs to bf16 (weights transposed so B operand is K-contiguous)
  k_f32_to_bf16<<<(MM * DD + 255) / 256, 256, 0, stream>>>(x, xb, MM * DD);
  {
    dim3 blk(16, 16);
    dim3 g1((TD + 15) / 16, (DD + 15) / 16);
    k_transpose_f32_to_bf16<<<g1, blk, 0, stream>>>(w_attn, wat, DD, TD);
    dim3 g2((DD + 15) / 16, (DD + 15) / 16);
    k_transpose_f32_to_bf16<<<g2, blk, 0, stream>>>(w_proj, wpt, DD, DD);
  }
  // 2) QKV projection: [8192,768] x [768,2304], 64x64 per block
  k_gemm_bf16<<<dim3(TD / 64, MM / 64), 128, 0, stream>>>(xb, wat, qkvb, MM, TD, DD);
  // 3) V transpose to [B,H,hd,L]
  k_build_vt<<<(BB * HH * HD * LL + 255) / 256, 256, 0, stream>>>(qkvb, vtb);
  // 4) causal flash attention
  k_flash_attn<<<dim3(LL / 64, HH, BB), 128, 0, stream>>>(qkvb, vtb, ao);
  // 5) output projection + bias: [8192,768] x [768,768] -> fp32
  k_gemm_f32_bias<<<dim3(DD / 64, MM / 64), 128, 0, stream>>>(ao, wpt, b_proj, outp, MM, DD, DD);
}
